// BailingMoeV25_73667279061185
// MI455X (gfx1250) — matrix operand order repaired and checked by reading, not compiled, not run on hardware
//
#include <hip/hip_runtime.h>
#include <hip/hip_bf16.h>

// ---------------- problem constants (match reference) ----------------
#define T_TOK 2048
#define HDIM  2048
#define NEXP  16
#define TOPK  4
#define IDIM  1024   // expert intermediate (== shared IS)
#define ROUTED_SCALE 2.5f

typedef __attribute__((ext_vector_type(16))) __bf16 v16bf;
typedef __attribute__((ext_vector_type(8)))  float  v8f;
typedef __attribute__((ext_vector_type(4)))  unsigned v4u;
typedef __attribute__((ext_vector_type(8)))  int      v8i;
typedef __attribute__((ext_vector_type(4)))  int      v4i;

#if defined(__has_builtin)
#if __has_builtin(__builtin_amdgcn_tensor_load_to_lds) && __has_builtin(__builtin_amdgcn_s_wait_tensorcnt)
#define HAVE_TDM 1
#endif
#endif

// ---------------- helpers ----------------
__device__ __forceinline__ unsigned short f2bf(float f) {
  unsigned u = __float_as_uint(f);
  unsigned r = u + 0x7FFFu + ((u >> 16) & 1u);   // round-to-nearest-even
  return (unsigned short)(r >> 16);
}
__device__ __forceinline__ unsigned pk2(float lo, float hi) {
  return (unsigned)f2bf(lo) | ((unsigned)f2bf(hi) << 16);
}
__device__ __forceinline__ float laneRead(float v, int srcLane) {
  // wave32 gather via LDS-permute unit (no LDS memory touched)
  return __int_as_float(__builtin_amdgcn_ds_bpermute(srcLane << 2, __float_as_int(v)));
}
__device__ __forceinline__ float fast_sigmoid(float z) {
  return __builtin_amdgcn_rcpf(1.f + __expf(-z));          // v_rcp_f32, not IEEE div chain
}

// ---------------- 1) router: sigmoid scores + grouped top-k -> C[T,E] ----------------
__global__ __launch_bounds__(128) void router_kernel(
    const float* __restrict__ x, const float* __restrict__ gw,
    const float* __restrict__ bias, float* __restrict__ C,
    int* __restrict__ idList, int* __restrict__ sharedCnt)
{
  int lane = threadIdx.x & 31;
  int wave = threadIdx.x >> 5;
  int t = blockIdx.x * 4 + wave;                 // grid is exact: T/4 blocks

  int e    = lane & 15;
  int part = lane >> 4;
  const float* xr = x  + (size_t)t * HDIM;
  const float* gr = gw + (size_t)e * HDIM;
  float acc = 0.f;
  for (int h = part; h < HDIM; h += 2)
    acc += xr[h] * gr[h];
  acc += laneRead(acc, lane ^ 16);               // lanes 0..15 now hold full logits

  float score = fast_sigmoid(acc);               // raw sigmoid score (weights)
  float sel   = score + bias[e];                 // selection score (biased)

  // gather all 16 values while EXEC is full (bpermute reads 0 from dead lanes)
  float sc[16], se[16];
  #pragma unroll
  for (int i = 0; i < 16; ++i) { sc[i] = laneRead(score, i); se[i] = laneRead(sel, i); }

  if (lane == 0) {
    // per-group (4 groups of 4) sum of top-2
    float gs[4];
    #pragma unroll
    for (int g = 0; g < 4; ++g) {
      float a = se[4*g], b = se[4*g+1], c = se[4*g+2], d = se[4*g+3];
      float h1 = fmaxf(a,b), l1 = fminf(a,b), h2 = fmaxf(c,d), l2 = fminf(c,d);
      float top = fmaxf(h1,h2);
      float sec = fmaxf(fminf(h1,h2), (h1 >= h2) ? l1 : l2);
      gs[g] = top + sec;
    }
    int g1 = 0; for (int g = 1; g < 4; ++g) if (gs[g] > gs[g1]) g1 = g;
    int g2 = -1; for (int g = 0; g < 4; ++g) { if (g == g1) continue; if (g2 < 0 || gs[g] > gs[g2]) g2 = g; }

    float m[16];
    #pragma unroll
    for (int i = 0; i < 16; ++i) { int g = i >> 2; m[i] = (g == g1 || g == g2) ? se[i] : -3.4e38f; }

    int ids[TOPK]; float wv[TOPK]; float wsum = 1e-20f;
    #pragma unroll
    for (int j = 0; j < TOPK; ++j) {
      int best = 0;
      for (int i = 1; i < 16; ++i) if (m[i] > m[best]) best = i;
      ids[j] = best; wv[j] = sc[best]; wsum += sc[best]; m[best] = -3.4e38f;
    }
    float rs = 1.f / wsum;
    float* Crow = C + (size_t)t * NEXP;
    #pragma unroll
    for (int i = 0; i < 16; ++i) Crow[i] = 0.f;
    #pragma unroll
    for (int j = 0; j < TOPK; ++j) Crow[ids[j]] = wv[j] * rs;

    idList[t] = t;                                // identity list for shared expert
  }
  if (blockIdx.x == 0 && threadIdx.x == 0) *sharedCnt = T_TOK;
}

// ---------------- 2) deterministic per-expert token gather ----------------
__global__ __launch_bounds__(32) void gather_kernel(
    const float* __restrict__ C, int* __restrict__ lists, int* __restrict__ counts)
{
  int e = blockIdx.x;
  int lane = threadIdx.x;
  int* list = lists + (size_t)e * T_TOK;
  int base = 0;
  for (int t0 = 0; t0 < T_TOK; t0 += 32) {
    int t = t0 + lane;
    bool hit = C[(size_t)t * NEXP + e] > 0.f;
    unsigned mk = __builtin_amdgcn_ballot_w32(hit);
    int pre = __popc(mk & ((1u << lane) - 1u));
    if (hit) list[base + pre] = t;                // token order preserved -> deterministic
    base += __popc(mk);
  }
  if (lane == 0) counts[e] = base;
}

// ---------------- 3) act = silu(X@w1^T) * (X@w3^T)  (bf16 WMMA, M=16, N=128/block) ----------------
__global__ __launch_bounds__(256) void act_kernel(
    const float* __restrict__ x, const float* __restrict__ w1, const float* __restrict__ w3,
    const int* __restrict__ list, const int* __restrict__ cntPtr,
    unsigned short* __restrict__ actBuf)          // bf16 raw, [rowInList][IDIM]
{
  int cnt = *cntPtr;
  int mTile = blockIdx.y;
  if (mTile * 16 >= cnt) return;
  int nBase = blockIdx.x * 128;
  int tid = threadIdx.x, lane = tid & 31, wave = tid >> 5;

  __shared__ int rowTok[16];
  __shared__ unsigned lA [16][17];                // X tile, bf16 pairs [m][kpair], +1 pad
  __shared__ unsigned lB1[128][17];               // w1 tile [n][kpair]
  __shared__ unsigned lB3[128][17];               // w3 tile [n][kpair]

  if (tid < 16) { int r = mTile * 16 + tid; rowTok[tid] = (r < cnt) ? list[r] : -1; }
  __syncthreads();

  int am = tid >> 4;                              // staging: row of X tile
  int ak = (tid & 15) * 2;                        // staging: k element
  int tokA = rowTok[am];
  const float* xp = (tokA >= 0) ? x + (size_t)tokA * HDIM + ak : nullptr;

  int wrow = tid >> 1;                            // staging: 0..127 (n within block tile)
  int wcol = (tid & 1) * 16;                      // 0 or 16 (k half)
  const float* w1p = w1 + (size_t)(nBase + wrow) * HDIM + wcol;
  const float* w3p = w3 + (size_t)(nBase + wrow) * HDIM + wcol;

  v8f accG = {0,0,0,0,0,0,0,0};
  v8f accU = {0,0,0,0,0,0,0,0};

  for (int k0 = 0; k0 < HDIM; k0 += 32) {
    float xa = 0.f, xb = 0.f;
    if (xp) { xa = xp[k0]; xb = xp[k0 + 1]; }
    lA[am][ak >> 1] = pk2(xa, xb);
    const float* p1 = w1p + k0;
    const float* p3 = w3p + k0;
    __builtin_prefetch(p1 + 32, 0, 0);            // global_prefetch_b8: next k-step
    __builtin_prefetch(p3 + 32, 0, 0);
    #pragma unroll
    for (int j = 0; j < 4; ++j) {
      float4 v1 = *(const float4*)(p1 + j * 4);
      float4 v3 = *(const float4*)(p3 + j * 4);
      lB1[wrow][(wcol >> 1) + j*2    ] = pk2(v1.x, v1.y);
      lB1[wrow][(wcol >> 1) + j*2 + 1] = pk2(v1.z, v1.w);
      lB3[wrow][(wcol >> 1) + j*2    ] = pk2(v3.x, v3.y);
      lB3[wrow][(wcol >> 1) + j*2 + 1] = pk2(v3.z, v3.w);
    }
    __syncthreads();

    int half = lane >> 4, mr = lane & 15;
    union { v16bf v; unsigned u[8]; } fa, fb1, fb3;
    #pragma unroll
    for (int i = 0; i < 8; ++i)                   // ISA 16-bit A 16x32 layout
      fa.u[i] = lA[mr][(i & 3) + ((i >> 2) << 3) + (half << 2)];
    int nr = wave * 16 + (lane & 15);
    #pragma unroll
    for (int i = 0; i < 8; ++i) {                 // ISA 16-bit B 32x16 layout
      fb1.u[i] = lB1[nr][(i & 3) + ((i >> 2) << 3) + (half << 2)];
      fb3.u[i] = lB3[nr][(i & 3) + ((i >> 2) << 3) + (half << 2)];
    }
    accG = __builtin_amdgcn_wmma_f32_16x16x32_bf16(false, fa.v, false, fb1.v, (short)0, accG, false, false);
    accU = __builtin_amdgcn_wmma_f32_16x16x32_bf16(false, fa.v, false, fb3.v, (short)0, accU, false, false);
    __syncthreads();
  }

  // epilogue: silu(g) * u -> bf16 act buffer (row = position in expert list)
  int half = lane >> 4, nc = lane & 15;
  #pragma unroll
  for (int i = 0; i < 8; ++i) {
    int mrow = i + (half << 3);                   // f32 C/D layout: VGPR i, lane-half -> M
    int r = mTile * 16 + mrow;
    if (r < cnt) {
      float g = accG[i], u = accU[i];
      float a = g * fast_sigmoid(g) * u;          // silu via v_rcp_f32
      actBuf[(size_t)r * IDIM + nBase + wave * 16 + nc] = f2bf(a);
    }
  }
}

// ---------------- 4) down proj: out[t] (+)= scale * C[t,e] * (act @ w2^T) ----------------
__global__ __launch_bounds__(256) void down_kernel(
    const unsigned short* __restrict__ actBuf, const float* __restrict__ w2,
    const int* __restrict__ list, const int* __restrict__ cntPtr,
    const float* __restrict__ Cmat, int eIdx, float scale, int accumulate,
    float* __restrict__ out)
{
  int cnt = *cntPtr;
  int mTile = blockIdx.y;
  if (mTile * 16 >= cnt) return;
  int hBase = blockIdx.x * 128;
  int tid = threadIdx.x, lane = tid & 31, wave = tid >> 5;

  __shared__ int rowTok[16];
  __shared__ unsigned lA[16][17];                 // A tile: TDM fills with pad(16dw,+1dw)
  __shared__ unsigned lB[128][17];

  if (tid < 16) { int r = mTile * 16 + tid; rowTok[tid] = (r < cnt) ? list[r] : -1; }

  const unsigned short* aBase = actBuf + (size_t)(mTile * 16) * IDIM;
  int rem = cnt - mTile * 16;                     // >0 here; TDM zero-fills rows >= rem

#ifdef HAVE_TDM
  unsigned ldsOff = (unsigned)(size_t)(void*)&lA[0][0];   // LDS aperture: addr[31:0]
#else
  int am = tid >> 4, ap = tid & 15;
  const unsigned short* aRow = aBase + (size_t)am * IDIM;
  bool aValid = am < rem;
#endif

  int wrow = tid >> 1;
  int wcol = (tid & 1) * 16;
  const float* w2p = w2 + (size_t)(hBase + wrow) * IDIM + wcol;

  v8f accD = {0,0,0,0,0,0,0,0};

  for (int k0 = 0; k0 < IDIM; k0 += 32) {
#ifdef HAVE_TDM
    if (tid < 32) {                               // one wave drives the TDM
      unsigned long long ga = (unsigned long long)(size_t)(const void*)(aBase) +
                              (unsigned long long)(k0 * 2);
      // D# group0: count=1 | lds_addr | global_addr[56:0] | type=2
      v4u g0 = { 1u, ldsOff, (unsigned)ga,
                 (unsigned)((ga >> 32) & 0x01FFFFFFull) | 0x80000000u };
      // D# group1: data_size=2B, pad_enable, pad every 16 DWORDs by 1 DWORD,
      //            tensor_dim0=IDIM, tensor_dim1=rem (OOB rows read as zero),
      //            tile 32x16, tensor_dim0_stride=IDIM
      v8i g1 = { (int)((1u << 16) | (1u << 20) | (3u << 22)),
                 (int)((unsigned)IDIM << 16),
                 (int)((unsigned)(rem > 0xFFFF ? 0xFFFF : rem) << 16),
                 (int)(32u << 16),
                 16, IDIM, 0, 0 };
      v4i gz = { 0, 0, 0, 0 };
#if defined(__clang_major__) && __clang_major__ >= 23
      v8i gz8 = { 0, 0, 0, 0, 0, 0, 0, 0 };
      __builtin_amdgcn_tensor_load_to_lds(g0, g1, gz, gz, gz8, 0);
#else
      __builtin_amdgcn_tensor_load_to_lds(g0, g1, gz, gz, 0);
#endif
    }
#else
    unsigned av = 0;
    if (aValid) av = *(const unsigned*)(aRow + k0 + ap * 2);   // already bf16 pairs
    lA[am][ap] = av;
#endif
    const float* p = w2p + k0;
    __builtin_prefetch(p + 32, 0, 0);
    #pragma unroll
    for (int j = 0; j < 4; ++j) {
      float4 v = *(const float4*)(p + j * 4);
      lB[wrow][(wcol >> 1) + j*2    ] = pk2(v.x, v.y);
      lB[wrow][(wcol >> 1) + j*2 + 1] = pk2(v.z, v.w);
    }
#ifdef HAVE_TDM
    if (tid < 32) __builtin_amdgcn_s_wait_tensorcnt(0);        // DMA done before barrier
#endif
    __syncthreads();

    int half = lane >> 4, mr = lane & 15;
    union { v16bf v; unsigned u[8]; } fa, fb;
    #pragma unroll
    for (int i = 0; i < 8; ++i)
      fa.u[i] = lA[mr][(i & 3) + ((i >> 2) << 3) + (half << 2)];
    int nr = wave * 16 + (lane & 15);
    #pragma unroll
    for (int i = 0; i < 8; ++i)
      fb.u[i] = lB[nr][(i & 3) + ((i >> 2) << 3) + (half << 2)];
    accD = __builtin_amdgcn_wmma_f32_16x16x32_bf16(false, fa.v, false, fb.v, (short)0, accD, false, false);
    __syncthreads();
  }

  int half = lane >> 4, nc = lane & 15;
  #pragma unroll
  for (int i = 0; i < 8; ++i) {
    int mrow = i + (half << 3);
    int r = mTile * 16 + mrow;
    if (r < cnt) {
      int t = rowTok[mrow];
      float w = scale;
      if (Cmat) w *= Cmat[(size_t)t * NEXP + eIdx];
      size_t o = (size_t)t * HDIM + hBase + wave * 16 + nc;
      float val = accD[i] * w;
      if (accumulate) out[o] += val; else out[o] = val;
    }
  }
}

// ---------------- launch ----------------
extern "C" void kernel_launch(void* const* d_in, const int* in_sizes, int n_in,
                              void* d_out, int out_size, void* d_ws, size_t ws_size,
                              hipStream_t stream)
{
  const float* x   = (const float*)d_in[0];
  const float* gw  = (const float*)d_in[1];
  const float* eb  = (const float*)d_in[2];
  const float* w1  = (const float*)d_in[3];
  const float* w3  = (const float*)d_in[4];
  const float* w2  = (const float*)d_in[5];
  const float* sw1 = (const float*)d_in[6];
  const float* sw3 = (const float*)d_in[7];
  const float* sw2 = (const float*)d_in[8];
  float* out = (float*)d_out;

  // workspace layout (~4.5 MB)
  char* ws = (char*)d_ws;
  float* C = (float*)ws;
  size_t off = (size_t)T_TOK * NEXP * sizeof(float);
  int* counts    = (int*)(ws + off); off += 64;
  int* sharedCnt = (int*)(ws + off); off += 64;
  int* lists     = (int*)(ws + off); off += (size_t)NEXP * T_TOK * sizeof(int);
  int* idList    = (int*)(ws + off); off += (size_t)T_TOK * sizeof(int);
  off = (off + 511) & ~(size_t)511;
  unsigned short* actBuf = (unsigned short*)(ws + off);   // T*I bf16 = 4 MB, reused serially

  router_kernel<<<T_TOK / 4, 128, 0, stream>>>(x, gw, eb, C, idList, sharedCnt);
  gather_kernel<<<NEXP, 32, 0, stream>>>(C, lists, counts);

  dim3 aGrid(IDIM / 128, T_TOK / 16);
  dim3 dGrid(HDIM / 128, T_TOK / 16);

  // shared expert first, in write mode (also initializes poisoned d_out)
  act_kernel <<<aGrid, 256, 0, stream>>>(x, sw1, sw3, idList, sharedCnt, actBuf);
  down_kernel<<<dGrid, 256, 0, stream>>>(actBuf, sw2, idList, sharedCnt, nullptr, 0, 1.0f, 0, out);

  // routed experts, serialized per-expert -> L2-resident weights, race-free accumulation
  for (int e = 0; e < NEXP; ++e) {
    act_kernel <<<aGrid, 256, 0, stream>>>(x, w1 + (size_t)e * IDIM * HDIM,
                                           w3 + (size_t)e * IDIM * HDIM,
                                           lists + (size_t)e * T_TOK, counts + e, actBuf);
    down_kernel<<<dGrid, 256, 0, stream>>>(actBuf, w2 + (size_t)e * HDIM * IDIM,
                                           lists + (size_t)e * T_TOK, counts + e,
                                           C, e, ROUTED_SCALE, 1, out);
  }
}
